// MultiHeadAttention_53102975648020
// MI455X (gfx1250) — matrix operand order repaired and checked by reading, not compiled, not run on hardware
//
#include <hip/hip_runtime.h>
#include <hip/hip_bf16.h>

// MHA for MI455X (gfx1250, wave32, WMMA bf16).
// B=4, L=2048, D=1024, H=16, DH=64.

#define B_  4
#define L_  2048
#define D_  1024
#define H_  16
#define DH_ 64
#define M_  (B_ * L_)   // 8192

typedef __attribute__((ext_vector_type(16))) __bf16 v16bf;
typedef __attribute__((ext_vector_type(8)))  __bf16 v8bf;
typedef __attribute__((ext_vector_type(8)))  float  v8f;

struct TrueT  { static constexpr bool value = true;  };
struct FalseT { static constexpr bool value = false; };

// A-fragment (16x32 bf16): lane = row M (both halves), K split:
// lanes 0-15 hold K {0..15}; lanes 16-31 hold K {16..31}.
__device__ __forceinline__ v16bf make_afrag(const __bf16* row, int hi) {
  v8bf a0 = *(const v8bf*)(row + 16 * hi);
  v8bf a1 = *(const v8bf*)(row + 16 * hi + 8);
  v16bf r;
#pragma unroll
  for (int i = 0; i < 8; ++i) { r[i] = a0[i]; r[i + 8] = a1[i]; }
  return r;
}

// B-fragment (32x16 bf16): lane holds ONE COLUMN n = lane&15; contraction
// K = 16*(lane>>4) .. +15 packed 2-per-VGPR ascending (k-major storage).
__device__ __forceinline__ v16bf load_bfrag(const __bf16* col_k16) {
  return *(const v16bf*)col_k16;
}

__device__ __forceinline__ v8f wmma_bf16(v16bf a, v16bf b, v8f c) {
  return __builtin_amdgcn_wmma_f32_16x16x32_bf16(
      /*neg_a=*/false, a, /*neg_b=*/false, b,
      /*c_mod=*/(short)0, c, /*reuse_a=*/false, /*reuse_b=*/false);
}

// Async memory->LDS copy of 32 contiguous bytes (2x GLOBAL_LOAD_ASYNC_TO_LDS_B128,
// ASYNCcnt-tracked; INST_OFFSET applies to both global and LDS addresses).
__device__ __forceinline__ void async_copy_32B(const void* gsrc, void* lds) {
  unsigned l = (unsigned)(unsigned long long)lds;  // LDS address (low 32 bits)
  asm volatile(
      "global_load_async_to_lds_b128 %0, %1, off\n\t"
      "global_load_async_to_lds_b128 %0, %1, off offset:16"
      :: "v"(l), "v"(gsrc) : "memory");
}
__device__ __forceinline__ void wait_async0() {
  asm volatile("s_wait_asynccnt 0x0" ::: "memory");
}

// ---------------------------------------------------------------------------
// GEMM: Y = A(8192x1024) * W^T(1024x1024).  128x64 block tile, 256 threads
// (8 waves as 4x2; each wave = 32x32 -> 2x2 WMMA accums = 4 WMMA/kstep).
// OUT_MODE: 0 = bf16 head-split [B,H,L,DH] (q,k); 1 = bf16 head-split
// [B,H,DH,L] (v, so attention needs no transpose); 2 = fp32 flat [M,D].
// ---------------------------------------------------------------------------
template <bool A_BF16, int OUT_MODE>
__global__ __launch_bounds__(256)
void gemm_wt_kernel(const void* __restrict__ Ap, const float* __restrict__ W,
                    void* __restrict__ Outp) {
  __shared__ __align__(32) __bf16 As[128][48];  // [m][k], row pitch 96B
  __shared__ __align__(32) __bf16 Bs[64][48];   // [n][k], row pitch 96B

  const int tid = threadIdx.x;
  const int lane = tid & 31, w = tid >> 5;
  const int lo = lane & 15, hi = lane >> 4;
  const int wr = w >> 1, wc = w & 1;            // wave grid 4 x 2
  const int m0 = blockIdx.y * 128, n0 = blockIdx.x * 64;

  v8f acc[2][2] = {{{}, {}}, {{}, {}}};

  const int arow = tid >> 1, ak = (tid & 1) * 16;  // A: 128 rows x 2 halves
  const int brow = tid >> 2, bk = (tid & 3) * 8;   // W: 64 rows x 4 chunks

  for (int k0 = 0; k0 < D_; k0 += 32) {
    __syncthreads();
    // ---- stage A tile 128x32 ----
    if constexpr (A_BF16) {
      const __bf16* A = (const __bf16*)Ap;
      async_copy_32B(A + (size_t)(m0 + arow) * D_ + k0 + ak, &As[arow][ak]);
    } else {
      const float* src = (const float*)Ap + (size_t)(m0 + arow) * D_ + k0 + ak;
      v16bf t;
#pragma unroll
      for (int i = 0; i < 16; ++i) t[i] = (__bf16)src[i];
      *(v16bf*)&As[arow][ak] = t;
    }
    // ---- stage W tile 64(n) x 32(k), natural [n][k] ----
    {
      const float* src = W + (size_t)(n0 + brow) * D_ + k0 + bk;
      v8bf t;
#pragma unroll
      for (int i = 0; i < 8; ++i) t[i] = (__bf16)src[i];
      *(v8bf*)&Bs[brow][bk] = t;
    }
    // prefetch next k-step (global_prefetch_b8)
    if (k0 + 32 < D_) {
      if constexpr (A_BF16)
        __builtin_prefetch((const __bf16*)Ap + (size_t)(m0 + arow) * D_ + k0 + 32 + ak, 0, 1);
      else
        __builtin_prefetch((const float*)Ap + (size_t)(m0 + arow) * D_ + k0 + 32 + ak, 0, 1);
      __builtin_prefetch(W + (size_t)(n0 + brow) * D_ + k0 + 32 + bk, 0, 1);
    }
    if constexpr (A_BF16) wait_async0();
    __syncthreads();

    v16bf af[2], bf[2];
#pragma unroll
    for (int rr = 0; rr < 2; ++rr)
      af[rr] = make_afrag(&As[wr * 32 + rr * 16 + lo][0], hi);
#pragma unroll
    for (int s = 0; s < 2; ++s)
      bf[s] = load_bfrag(&Bs[wc * 32 + s * 16 + lo][16 * hi]);
#pragma unroll
    for (int rr = 0; rr < 2; ++rr)
#pragma unroll
      for (int s = 0; s < 2; ++s)
        acc[rr][s] = wmma_bf16(af[rr], bf[s], acc[rr][s]);
  }

  // ---- epilogue: C layout -> row = 8*hi + j, col = lo ----
#pragma unroll
  for (int rr = 0; rr < 2; ++rr) {
#pragma unroll
    for (int j = 0; j < 8; ++j) {
      const int mg = m0 + wr * 32 + rr * 16 + 8 * hi + j;
#pragma unroll
      for (int s = 0; s < 2; ++s) {
        const int ng  = n0 + wc * 32 + s * 16 + lo;
        const float v = acc[rr][s][j];
        if constexpr (OUT_MODE == 0) {
          const int b = mg >> 11, l = mg & (L_ - 1);
          const int h = ng >> 6, dh = ng & (DH_ - 1);
          ((__bf16*)Outp)[(((size_t)b * H_ + h) * L_ + l) * DH_ + dh] = (__bf16)v;
        } else if constexpr (OUT_MODE == 1) {
          const int b = mg >> 11, l = mg & (L_ - 1);
          const int h = ng >> 6, dh = ng & (DH_ - 1);
          ((__bf16*)Outp)[(((size_t)b * H_ + h) * DH_ + dh) * L_ + l] = (__bf16)v;
        } else {
          ((float*)Outp)[(size_t)mg * D_ + ng] = v;
        }
      }
    }
  }
}

// ---------------------------------------------------------------------------
// Causal flash attention.  128 threads = 4 waves; wave owns a 16-row Q tile;
// block shares K ([key][dh]) and V ([dh][key], pre-transposed in global) tiles
// of 32 keys, staged with async global->LDS b128 copies.
// Softmax in exp2 domain; row-sums computed by WMMA against an all-ones
// B operand (replaces the shuffle-add reduction); full (unmasked) key blocks
// skip the causal compare/select entirely.
// ---------------------------------------------------------------------------
__global__ __launch_bounds__(128)
void attn_kernel(const __bf16* __restrict__ qh, const __bf16* __restrict__ kh,
                 const __bf16* __restrict__ vt, __bf16* __restrict__ attn) {
  __shared__ __align__(32) __bf16 Ks[32][80];      // [key][dh], pitch 160B
  __shared__ __align__(32) __bf16 Vt[64][48];      // [dh][key], pitch 96B
  __shared__ __align__(32) __bf16 Ps[4][16][48];   // per-wave P [q][key]

  const int tid = threadIdx.x;
  const int lane = tid & 31, w = tid >> 5;
  const int lo = lane & 15, hi = lane >> 4;
  const int bh = blockIdx.x >> 5;   // b*H + h
  const int qt = blockIdx.x & 31;
  const int q0 = qt * 64 + w * 16;
  const size_t base = (size_t)bh * L_ * DH_;   // per-head element offset

  v16bf qf[2];
#pragma unroll
  for (int c = 0; c < 2; ++c)
    qf[c] = make_afrag(qh + base + (size_t)(q0 + lo) * DH_ + c * 32, hi);

  v16bf ones;
#pragma unroll
  for (int i = 0; i < 16; ++i) ones[i] = (__bf16)1.0f;

  float m_r[8], l_r[8];
  v8f o0 = {}, o1 = {}, o2 = {}, o3 = {};
#pragma unroll
  for (int j = 0; j < 8; ++j) { m_r[j] = -3.0e38f; l_r[j] = 0.f; }

  const int kp   = tid >> 2;          // 0..31  (K staging row)
  const int dh0  = (tid & 3) * 16;    // 0,16,32,48
  const int dhr  = tid >> 1;          // 0..63  (V staging row)
  const int kv0  = (tid & 1) * 16;    // 0,16
  const float SC = 0.125f * 1.44269504f;  // 1/sqrt(DH) * log2(e)
  const int kbN  = 2 * qt + 2;

  for (int kb = 0; kb < kbN; ++kb) {
    const int ks = kb * 32;
    __syncthreads();
    // ---- async stage: K 32x64 natural, V 64x32 (already dh-major) ----
    async_copy_32B(kh + base + (size_t)(ks + kp) * DH_ + dh0, &Ks[kp][dh0]);
    async_copy_32B(vt + base + (size_t)dhr * L_ + ks + kv0,   &Vt[dhr][kv0]);
    wait_async0();
    __syncthreads();

    // ---- S = Q * K^T : per accum s, lane column = key ks+s*16+lo ----
    v8f s0 = {}, s1 = {};
#pragma unroll
    for (int c = 0; c < 2; ++c) {
      v16bf bk0 = load_bfrag(&Ks[0 * 16 + lo][c * 32 + 16 * hi]);
      v16bf bk1 = load_bfrag(&Ks[1 * 16 + lo][c * 32 + 16 * hi]);
      s0 = wmma_bf16(qf[c], bk0, s0);
      s1 = wmma_bf16(qf[c], bk1, s1);
    }

    // ---- online softmax pass 1: max/exp/scale, write P (exp2 domain) ----
    float alpha[8];
    auto pass1 = [&](auto Mv) {
#pragma unroll
      for (int j = 0; j < 8; ++j) {
        float v0, v1;
        if constexpr (decltype(Mv)::value) {
          const int qg = q0 + 8 * hi + j;
          v0 = (ks + lo      <= qg) ? s0[j] * SC : -3.0e38f;
          v1 = (ks + 16 + lo <= qg) ? s1[j] * SC : -3.0e38f;
        } else {
          v0 = s0[j] * SC;
          v1 = s1[j] * SC;
        }
        float rm = fmaxf(v0, v1);
#pragma unroll
        for (int msk = 1; msk < 16; msk <<= 1)
          rm = fmaxf(rm, __shfl_xor(rm, msk, 32));
        const float mn = fmaxf(m_r[j], rm);
        alpha[j] = exp2f(m_r[j] - mn);
        m_r[j]   = mn;
        const int r = 8 * hi + j;
        Ps[w][r][lo]      = (__bf16)exp2f(v0 - mn);
        Ps[w][r][16 + lo] = (__bf16)exp2f(v1 - mn);
        o0[j] *= alpha[j]; o1[j] *= alpha[j];
        o2[j] *= alpha[j]; o3[j] *= alpha[j];
      }
    };
    if (ks + 31 <= q0) pass1(FalseT{}); else pass1(TrueT{});

    // Same-wave LDS RAW is in-order; no barrier needed for the per-wave tile.
    v16bf pa = make_afrag(&Ps[w][lo][0], hi);

    // ---- row sums via WMMA against all-ones B (replaces shuffle-add) ----
    v8f rs = {};
    rs = wmma_bf16(pa, ones, rs);
#pragma unroll
    for (int j = 0; j < 8; ++j) l_r[j] = l_r[j] * alpha[j] + rs[j];

    // ---- O += P * V ----
    v16bf bv0 = load_bfrag(&Vt[0 * 16 + lo][16 * hi]);
    v16bf bv1 = load_bfrag(&Vt[1 * 16 + lo][16 * hi]);
    v16bf bv2 = load_bfrag(&Vt[2 * 16 + lo][16 * hi]);
    v16bf bv3 = load_bfrag(&Vt[3 * 16 + lo][16 * hi]);
    o0 = wmma_bf16(pa, bv0, o0);
    o1 = wmma_bf16(pa, bv1, o1);
    o2 = wmma_bf16(pa, bv2, o2);
    o3 = wmma_bf16(pa, bv3, o3);
  }

  // ---- normalize and write attn as [B, L, H*DH] (== final GEMM A layout) ----
  const int b = bh >> 4, h = bh & 15;
#pragma unroll
  for (int j = 0; j < 8; ++j) {
    const int qg = q0 + 8 * hi + j;
    const float inv = 1.0f / l_r[j];
    __bf16* dst = attn + (((size_t)b * L_ + qg) * H_ + h) * DH_;
    dst[lo]      = (__bf16)(o0[j] * inv);
    dst[16 + lo] = (__bf16)(o1[j] * inv);
    dst[32 + lo] = (__bf16)(o2[j] * inv);
    dst[48 + lo] = (__bf16)(o3[j] * inv);
  }
}

extern "C" void kernel_launch(void* const* d_in, const int* in_sizes, int n_in,
                              void* d_out, int out_size, void* d_ws, size_t ws_size,
                              hipStream_t stream) {
  (void)in_sizes; (void)n_in; (void)out_size; (void)ws_size;
  const float* q   = (const float*)d_in[0];
  const float* k   = (const float*)d_in[1];
  const float* v   = (const float*)d_in[2];
  // d_in[3] = mask: causal tril, handled analytically in attn_kernel.
  const float* W_q = (const float*)d_in[4];
  const float* W_k = (const float*)d_in[5];
  const float* W_v = (const float*)d_in[6];
  const float* W_o = (const float*)d_in[7];

  const size_t SZ = (size_t)M_ * D_;   // elements per bf16 tensor
  __bf16* qh   = (__bf16*)d_ws;        // [B,H,L,DH]
  __bf16* kh   = qh + SZ;              // [B,H,L,DH]
  __bf16* vt   = kh + SZ;              // [B,H,DH,L]  (pre-transposed)
  __bf16* attn = vt + SZ;              // [B,L,H*DH]

  dim3 gg(D_ / 64, M_ / 128);  // (16, 64)
  gemm_wt_kernel<false, 0><<<gg, 256, 0, stream>>>(q, W_q, qh);
  gemm_wt_kernel<false, 0><<<gg, 256, 0, stream>>>(k, W_k, kh);
  gemm_wt_kernel<false, 1><<<gg, 256, 0, stream>>>(v, W_v, vt);

  attn_kernel<<<(B_ * H_) * (L_ / 64), 128, 0, stream>>>(qh, kh, vt, attn);

  gemm_wt_kernel<true, 2><<<gg, 256, 0, stream>>>(attn, W_o, d_out);
}
